// Feat_encoder_84456236908567
// MI455X (gfx1250) — matrix operand order repaired and checked by reading, not compiled, not run on hardware
//
#include <hip/hip_runtime.h>

#define KNEI 32
#define DIMS 128
#define ROWSTRIDE 132   // floats; padded (132*4B = 528B, multiple of 16B)
#define WPB 2           // waves (nodes) per block -> 34 KB LDS per block
#define LEAKY_SLOPE 0.01f

typedef __attribute__((ext_vector_type(16))) _Float16 v16h;
typedef __attribute__((ext_vector_type(8)))  float    v8f;

__device__ __forceinline__ float wred_sum(float v) {
#pragma unroll
  for (int m = 16; m >= 1; m >>= 1) v += __shfl_xor(v, m, 32);
  return v;
}
__device__ __forceinline__ float wred_max(float v) {
#pragma unroll
  for (int m = 16; m >= 1; m >>= 1) v = fmaxf(v, __shfl_xor(v, m, 32));
  return v;
}

__global__ __launch_bounds__(WPB * 32) void gat_fused_kernel(
    const float* __restrict__ feat,
    const int*   __restrict__ nei,
    const float* __restrict__ att,
    float*       __restrict__ out,
    int nnodes)
{
  __shared__ __align__(16) float E[WPB][KNEI * ROWSTRIDE];
  __shared__ float Wt[WPB][KNEI];

  const int lane = threadIdx.x & 31;
  const int wv   = threadIdx.x >> 5;
  const int n    = blockIdx.x * WPB + wv;
  if (n >= nnodes) return;   // whole wave exits together (EXEC stays all-1s for WMMA)

  // ---- per-lane neighbor row pointer (lane k owns neighbor k) ----
  const int idx = nei[n * KNEI + lane];
  const float* grow = feat + (size_t)idx * DIMS;
  float* erow = &E[wv][lane * ROWSTRIDE];

  // ---- stage the 32x128 f32 gather into LDS via gfx1250 async DMA ----
  // LDS dest is the vdst VGPR plus the inst offset; generic LDS ptr low 32
  // bits are the LDS byte offset (aperture is in bits [63:32]).
  {
    const unsigned long long gaddr = (unsigned long long)(uintptr_t)grow;
    const unsigned int loff = (unsigned int)(uintptr_t)erow;
#pragma unroll
    for (int j = 0; j < DIMS / 4; ++j) {   // 32 async b128 copies in flight
      asm volatile("global_load_async_to_lds_b128 %0, %1, off offset:%2"
                   :: "v"(loff), "v"(gaddr), "n"(16 * j)
                   : "memory");
    }
  }

  // ---- overlap: score_ref = dot(feat[n], a_ref), 4 dims per lane ----
  const float4 fr = *(const float4*)(feat + (size_t)n * DIMS + lane * 4);
  const float4 ar = *(const float4*)(att + lane * 4);
  float sref = fr.x * ar.x + fr.y * ar.y + fr.z * ar.z + fr.w * ar.w;
  sref = wred_sum(sref);

  // ---- overlap: build B fragments (a_nei chunk broadcast to all 16 cols) ----
  // Dense f16 B (32x16) layout: VGPR v half h -> K = 2v+h (v<4) or 2v+h+8 (v>=4), +8 for lanes 16-31, N = lane%16.
  // B[k][j] = a_nei[32c + k] for all j  => lane needs floats {kb..kb+7, 16+kb..16+kb+7}.
  const float* anei = att + DIMS;
  const int boff = (lane < 16) ? 0 : 8;
  v16h Bf[4];
#pragma unroll
  for (int c = 0; c < 4; ++c) {
    const float* bp = anei + 32 * c + boff;
#pragma unroll
    for (int q = 0; q < 4; ++q) {
      float4 b4 = *(const float4*)(bp + 4 * q + (q >= 2 ? 8 : 0));
      Bf[c][4 * q + 0] = (_Float16)b4.x;
      Bf[c][4 * q + 1] = (_Float16)b4.y;
      Bf[c][4 * q + 2] = (_Float16)b4.z;
      Bf[c][4 * q + 3] = (_Float16)b4.w;
    }
  }

  // ---- wait for the async gather before touching the LDS tile ----
  asm volatile("s_wait_asynccnt 0" ::: "memory");

  // ---- neighbor scores via v_wmma_f32_16x16x32_f16 ----
  // A (16x32 f16) layout: m = lane&15 is row M; lanes<16 hold K {0..7,16..23},
  // lanes>=16 hold K {8..15,24..31}; pair p -> k0 = (p<4 ? kb+2p : 16+kb+2(p-4)).
  const int m  = lane & 15;
  const int kb = (lane < 16) ? 0 : 8;
  v8f acc0 = {}; v8f acc1 = {};
#pragma unroll
  for (int c = 0; c < 4; ++c) {          // K chunks of 32 feature dims
#pragma unroll
    for (int t = 0; t < 2; ++t) {        // neighbor tiles 0..15 / 16..31
      const float* rowp = &E[wv][(t * 16 + m) * ROWSTRIDE + 32 * c];
      v16h A;
#pragma unroll
      for (int p = 0; p < 8; ++p) {
        const int k0 = (p < 4) ? (kb + 2 * p) : (16 + kb + 2 * (p - 4));
        float2 f2 = *(const float2*)(rowp + k0);
        A[2 * p + 0] = (_Float16)f2.x;
        A[2 * p + 1] = (_Float16)f2.y;
      }
      if (t == 0)
        acc0 = __builtin_amdgcn_wmma_f32_16x16x32_f16(false, A, false, Bf[c],
                                                      (short)0, acc0, false, false);
      else
        acc1 = __builtin_amdgcn_wmma_f32_16x16x32_f16(false, A, false, Bf[c],
                                                      (short)0, acc1, false, false);
    }
  }

  // ---- spill the 32 scores (C/D layout: VGPR r -> M=r (lanes<16) / r+8 (lanes>=16)) ----
  if (lane == 0 || lane == 16) {
    const int base = (lane == 0) ? 0 : 8;
#pragma unroll
    for (int r = 0; r < 8; ++r) {
      Wt[wv][base + r]      = acc0[r];
      Wt[wv][16 + base + r] = acc1[r];
    }
  }
  __builtin_amdgcn_wave_barrier();

  // ---- leaky-relu + softmax over K=32 (lane k owns score k) ----
  float s = sref + Wt[wv][lane];
  s = (s >= 0.0f) ? s : LEAKY_SLOPE * s;
  const float mx = wred_max(s);
  const float e  = __expf(s - mx);
  const float sm = wred_sum(e);
  Wt[wv][lane] = e / sm;
  __builtin_amdgcn_wave_barrier();

  // ---- weighted combine: out[d] = sum_k w_k * E[k][d], 4 dims per lane ----
  float ox = 0.f, oy = 0.f, oz = 0.f, ow = 0.f;
#pragma unroll 8
  for (int k = 0; k < KNEI; ++k) {
    const float wk = Wt[wv][k];                                // broadcast read
    const float4 e4 = *(const float4*)(&E[wv][k * ROWSTRIDE + lane * 4]);
    ox = fmaf(wk, e4.x, ox);
    oy = fmaf(wk, e4.y, oy);
    oz = fmaf(wk, e4.z, oz);
    ow = fmaf(wk, e4.w, ow);
  }

  // ---- ELU + coalesced b128 store ----
  float4 o;
  o.x = (ox >= 0.f) ? ox : expm1f(ox);
  o.y = (oy >= 0.f) ? oy : expm1f(oy);
  o.z = (oz >= 0.f) ? oz : expm1f(oz);
  o.w = (ow >= 0.f) ? ow : expm1f(ow);
  *(float4*)(out + (size_t)n * DIMS + lane * 4) = o;
}

extern "C" void kernel_launch(void* const* d_in, const int* in_sizes, int n_in,
                              void* d_out, int out_size, void* d_ws, size_t ws_size,
                              hipStream_t stream) {
  (void)n_in; (void)out_size; (void)d_ws; (void)ws_size;
  const float* feat = (const float*)d_in[0];
  const int*   nei  = (const int*)d_in[1];
  const float* att  = (const float*)d_in[2];
  float* out = (float*)d_out;
  const int nnodes = in_sizes[1] / KNEI;   // 50000
  const int blocks = (nnodes + WPB - 1) / WPB;
  gat_fused_kernel<<<blocks, WPB * 32, 0, stream>>>(feat, nei, att, out, nnodes);
}
